// Test_1_77644418777135
// MI455X (gfx1250) — matrix operand order repaired and checked by reading, not compiled, not run on hardware
//
#include <hip/hip_runtime.h>
#include <hip/hip_bf16.h>

// ---------------------------------------------------------------------------
// Mamba stack on gfx1250: bf16 WMMA GEMMs (async-LDS weight staging) +
// wave32 selective scan.
// ---------------------------------------------------------------------------

typedef __bf16 bf16_t;
typedef __attribute__((ext_vector_type(16))) __bf16 v16bf;
typedef __attribute__((ext_vector_type(8)))  __bf16 v8bf;
typedef __attribute__((ext_vector_type(8)))  float  v8f;
typedef __attribute__((ext_vector_type(4)))  int    v4i_t;
// exact pointer types for the async-LDS builtin (from hipcc's diagnostic):
typedef __attribute__((address_space(1))) v4i_t* g_v4i_p;   // global int4*
typedef __attribute__((address_space(3))) v4i_t* l_v4i_p;   // LDS    int4*

#define T_TOKENS 8192      // 4 * 2048
#define SEQ      2048
#define DM       256
#define DI       512
#define NST      16
#define DTR      16

#if defined(__AMDGCN__) && __has_builtin(__builtin_amdgcn_global_load_async_to_lds_b128) && __has_builtin(__builtin_amdgcn_s_wait_asynccnt)
#define USE_ASYNC_LDS 1
#else
#define USE_ASYNC_LDS 0
#endif

// ----------------------------- weight pack (f32 KxN -> bf16 NxK) -----------
__global__ void packT_kernel(const float* __restrict__ W, bf16_t* __restrict__ Wt,
                             int K, int N) {
  int idx = blockIdx.x * blockDim.x + threadIdx.x;
  if (idx >= K * N) return;
  int n = idx / K;
  int k = idx - n * K;
  Wt[(size_t)n * K + k] = (bf16_t)W[(size_t)k * N + n];
}

// ----------------------------- WMMA GEMM -----------------------------------
// C[M,N] = A[M,K](f32) @ Wt[N,K](bf16, transposed weight) + bias
// block tile 64x64, 128 threads = 4 waves; each wave: 16 rows x 64 cols.
// Weight tile goes to LDS via GLOBAL_LOAD_ASYNC_TO_LDS_B128 (ASYNCcnt path).
#define BM 64
#define BN 64
#define BK 32

__global__ __launch_bounds__(128) void gemm_bf16_wmma(
    const float* __restrict__ A, int lda,
    const bf16_t* __restrict__ Wt,
    const float* __restrict__ bias,
    float* __restrict__ C, int ldc,
    int M, int N, int K) {
  __shared__ bf16_t As[BM][BK + 8];
  __shared__ bf16_t Bs[BN][BK + 8];

  const int tid  = threadIdx.x;
  const int lane = tid & 31;
  const int wave = tid >> 5;
  const int l16  = lane & 15;
  const int half = lane >> 4;          // 0: lanes 0-15, 1: lanes 16-31
  const int bm = blockIdx.x * BM;
  const int bn = blockIdx.y * BN;

  // A staging: thread owns half a row (16 contiguous floats)
  const int rA = tid >> 1;
  const int cA = (tid & 1) * 16;

  v8f acc[4];
#pragma unroll
  for (int ct = 0; ct < 4; ++ct)
#pragma unroll
    for (int i = 0; i < 8; ++i) acc[ct][i] = 0.f;

  for (int k0 = 0; k0 < K; k0 += BK) {
    // ---- stage A tile: 4x float4 loads -> pack -> 2x 16B LDS stores ----
    {
      const float4* src =
          (const float4*)&A[(size_t)(bm + rA) * lda + (k0 + cA)];
      float4 f0 = src[0], f1 = src[1], f2 = src[2], f3 = src[3];
      v8bf p0, p1;
      p0[0] = (bf16_t)f0.x; p0[1] = (bf16_t)f0.y;
      p0[2] = (bf16_t)f0.z; p0[3] = (bf16_t)f0.w;
      p0[4] = (bf16_t)f1.x; p0[5] = (bf16_t)f1.y;
      p0[6] = (bf16_t)f1.z; p0[7] = (bf16_t)f1.w;
      p1[0] = (bf16_t)f2.x; p1[1] = (bf16_t)f2.y;
      p1[2] = (bf16_t)f2.z; p1[3] = (bf16_t)f2.w;
      p1[4] = (bf16_t)f3.x; p1[5] = (bf16_t)f3.y;
      p1[6] = (bf16_t)f3.z; p1[7] = (bf16_t)f3.w;
      *(v8bf*)&As[rA][cA]     = p0;
      *(v8bf*)&As[rA][cA + 8] = p1;
    }
    // ---- stage weight tile: 64 rows x 64B = 256 x 16B chunks ----
    // (no N-edge zero-fill: out-of-range rows only feed masked-out columns;
    //  workspace has a pad region so reads stay in-bounds of d_ws)
#pragma unroll
    for (int i = 0; i < 2; ++i) {
      int q   = tid + i * 128;
      int row = q >> 2;
      int col = (q & 3) * 8;                 // bf16 elements
      const bf16_t* gsrc = &Wt[(size_t)(bn + row) * K + (k0 + col)];
      bf16_t*       ldst = &Bs[row][col];
#if USE_ASYNC_LDS
      __builtin_amdgcn_global_load_async_to_lds_b128(
          (g_v4i_p)(void*)gsrc, (l_v4i_p)(void*)ldst, 0, 0);
#else
      *(uint4*)ldst = *(const uint4*)gsrc;
#endif
    }
    if (k0 + BK < K) {
      // hint next A tile into cache (global_prefetch_b8)
      __builtin_prefetch(&A[(size_t)(bm + rA) * lda + (k0 + BK + cA)], 0, 0);
    }
#if USE_ASYNC_LDS
    __builtin_amdgcn_s_wait_asynccnt(0);
#endif
    __syncthreads();

    // ---- fragments ----
    // A: lane<16 holds row M=l16 with K {0..7,16..23}; lane>=16 K {8..15,24..31}
    const int arow = wave * 16 + l16;
    const int kbA  = half * 8;
    v8bf alo = *(const v8bf*)&As[arow][kbA];
    v8bf ahi = *(const v8bf*)&As[arow][kbA + 16];
    v16bf afrag = __builtin_shufflevector(alo, ahi, 0, 1, 2, 3, 4, 5, 6, 7,
                                          8, 9, 10, 11, 12, 13, 14, 15);
    // B: lane<16 holds K {0..7,16..23} of its column; lane>=16 K {8..15,24..31}
    const int kbB = half * 8;
#pragma unroll
    for (int ct = 0; ct < 4; ++ct) {
      const int bcol = ct * 16 + l16;
      v8bf blo = *(const v8bf*)&Bs[bcol][kbB];
      v8bf bhi = *(const v8bf*)&Bs[bcol][kbB + 16];
      v16bf bfrag = __builtin_shufflevector(blo, bhi, 0, 1, 2, 3, 4, 5, 6, 7,
                                            8, 9, 10, 11, 12, 13, 14, 15);
      acc[ct] = __builtin_amdgcn_wmma_f32_16x16x32_bf16(
          false, afrag, false, bfrag, (short)0, acc[ct], false, false);
    }
    __syncthreads();
  }

  // writeback: VGPR i -> M = wave*16 + half*8 + i, N = ct*16 + l16
#pragma unroll
  for (int ct = 0; ct < 4; ++ct) {
    int n = bn + ct * 16 + l16;
    if (n >= N) continue;
    float bv = bias ? bias[n] : 0.f;
#pragma unroll
    for (int i = 0; i < 8; ++i) {
      int m = bm + wave * 16 + half * 8 + i;
      C[(size_t)m * ldc + n] = acc[ct][i] + bv;
    }
  }
}

// ----------------------------- rmsnorm (optional residual add) --------------
template <int D>
__global__ __launch_bounds__(256) void rmsnorm_kernel(
    const float* __restrict__ X, int ldx,
    const float* __restrict__ Add, int ldadd,
    float* __restrict__ Out, int ldo) {
  __shared__ float red[8];
  const int t = blockIdx.x;
  float vals[D / 256];
  float ss = 0.f;
#pragma unroll
  for (int j = 0; j < D / 256; ++j) {
    int c = threadIdx.x + j * 256;
    float v = X[(size_t)t * ldx + c];
    if (Add) v += Add[(size_t)t * ldadd + c];
    vals[j] = v;
    ss += v * v;
  }
  for (int off = 16; off > 0; off >>= 1) ss += __shfl_xor(ss, off, 32);
  if ((threadIdx.x & 31) == 0) red[threadIdx.x >> 5] = ss;
  __syncthreads();
  float tot = 0.f;
#pragma unroll
  for (int w = 0; w < 8; ++w) tot += red[w];
  const float inv = rsqrtf(tot / (float)D + 1e-5f);
#pragma unroll
  for (int j = 0; j < D / 256; ++j)
    Out[(size_t)t * ldo + threadIdx.x + j * 256] = vals[j] * inv;
}

// ----------------------------- depthwise conv (SAME, W=4) + silu ------------
__global__ __launch_bounds__(256) void conv_silu_kernel(
    const float* __restrict__ xp, int ldx,
    const float* __restrict__ cw, const float* __restrict__ cb,
    float* __restrict__ U) {
  int idx = blockIdx.x * blockDim.x + threadIdx.x;
  if (idx >= T_TOKENS * DI) return;
  int d = idx & (DI - 1);
  int t = idx >> 9;
  int l = t & (SEQ - 1);
  int bbase = t - l;
  float acc = cb[d];
#pragma unroll
  for (int w = 0; w < 4; ++w) {
    int ls = l - 1 + w;                       // SAME pad: lo=1, hi=2
    if (ls >= 0 && ls < SEQ)
      acc += xp[(size_t)(bbase + ls) * ldx + d] * cw[w * DI + d];
  }
  U[idx] = acc / (1.f + __expf(-acc));        // silu
}

// ----------------------------- delta = softplus(proj_dt @ dt_w + b) ---------
__global__ __launch_bounds__(256) void dtproj_kernel(
    const float* __restrict__ proj,
    const float* __restrict__ dtw, const float* __restrict__ dtb,
    float* __restrict__ Delta) {
  int idx = blockIdx.x * blockDim.x + threadIdx.x;
  if (idx >= T_TOKENS * DI) return;
  int d = idx & (DI - 1);
  int t = idx >> 9;
  const float* pr = proj + (size_t)t * 48;
  float acc = dtb[d];
#pragma unroll
  for (int r = 0; r < DTR; ++r) acc += pr[r] * dtw[r * DI + d];
  Delta[idx] = (acc > 20.f) ? acc : log1pf(__expf(acc));
}

// ----------------------------- selective scan (wave32, 16-lane reduce) ------
// thread = (b, d, n). 16 lanes per d; butterfly-reduce partial y over n.
__global__ __launch_bounds__(256) void scan_kernel(
    const float* __restrict__ U,        // (T,512)
    const float* __restrict__ Delta,    // (T,512)
    const float* __restrict__ proj,     // (T,48): B at +16, C at +32
    const float* __restrict__ Alog,     // (512,16)
    const float* __restrict__ Dp,       // (512)
    const float* __restrict__ Res, int ldr,  // gate input (stride 1024)
    float* __restrict__ Y) {            // (T,512)
  const int b = blockIdx.x >> 5;                    // 32 blocks per batch
  const int dblk = (blockIdx.x & 31) * 16;
  const int n = threadIdx.x & 15;
  const int d = dblk + (threadIdx.x >> 4);
  const float Ac = -__expf(Alog[d * NST + n]);
  const float Dv = Dp[d];
  float s = 0.f;
  const int tbase = b * SEQ;
  for (int l = 0; l < SEQ; ++l) {
    const int t = tbase + l;
    const float dl = Delta[(size_t)t * DI + d];
    const float uv = U[(size_t)t * DI + d];
    const float Bv = proj[(size_t)t * 48 + DTR + n];
    const float Cv = proj[(size_t)t * 48 + DTR + NST + n];
    s = __expf(dl * Ac) * s + dl * Bv * uv;
    float p = s * Cv;
    p += __shfl_xor(p, 1, 32);
    p += __shfl_xor(p, 2, 32);
    p += __shfl_xor(p, 4, 32);
    p += __shfl_xor(p, 8, 32);
    if (n == 0) {
      float r = Res[(size_t)t * ldr + d];
      float g = r / (1.f + __expf(-r));             // silu gate
      Y[(size_t)t * DI + d] = (p + uv * Dv) * g;
    }
  }
}

// ----------------------------- softmax(256) then layernorm ------------------
__global__ __launch_bounds__(256) void smax_ln_kernel(
    const float* __restrict__ O,
    const float* __restrict__ gs, const float* __restrict__ gb,
    float* __restrict__ H) {
  __shared__ float red[8];
  const int t = blockIdx.x;
  const int c = threadIdx.x;
  float v = O[(size_t)t * DM + c];

  float m = v;
  for (int off = 16; off > 0; off >>= 1) m = fmaxf(m, __shfl_xor(m, off, 32));
  if ((c & 31) == 0) red[c >> 5] = m;
  __syncthreads();
  float mm = red[0];
#pragma unroll
  for (int w = 1; w < 8; ++w) mm = fmaxf(mm, red[w]);
  __syncthreads();

  float e = __expf(v - mm);
  float ssum = e;
  for (int off = 16; off > 0; off >>= 1) ssum += __shfl_xor(ssum, off, 32);
  if ((c & 31) == 0) red[c >> 5] = ssum;
  __syncthreads();
  float tot = 0.f;
#pragma unroll
  for (int w = 0; w < 8; ++w) tot += red[w];
  __syncthreads();
  const float p = e / tot;

  float mus = p;
  for (int off = 16; off > 0; off >>= 1) mus += __shfl_xor(mus, off, 32);
  if ((c & 31) == 0) red[c >> 5] = mus;
  __syncthreads();
  float mu = 0.f;
#pragma unroll
  for (int w = 0; w < 8; ++w) mu += red[w];
  mu *= (1.f / DM);
  __syncthreads();

  float dv = (p - mu) * (p - mu);
  for (int off = 16; off > 0; off >>= 1) dv += __shfl_xor(dv, off, 32);
  if ((c & 31) == 0) red[c >> 5] = dv;
  __syncthreads();
  float var = 0.f;
#pragma unroll
  for (int w = 0; w < 8; ++w) var += red[w];
  var *= (1.f / DM);

  H[(size_t)t * DM + c] = (p - mu) * rsqrtf(var + 1e-6f) * gs[c] + gb[c];
}

// ----------------------------- head: (T,256)@(256,17) -----------------------
__global__ __launch_bounds__(256) void head_kernel(
    const float* __restrict__ H,
    const float* __restrict__ Wo, const float* __restrict__ bo,
    float* __restrict__ Out) {
  int idx = blockIdx.x * blockDim.x + threadIdx.x;
  if (idx >= T_TOKENS * 17) return;
  int a = idx % 17;
  int t = idx / 17;
  float acc = bo[a];
  const float* h = H + (size_t)t * DM;
#pragma unroll 4
  for (int j = 0; j < DM; ++j) acc += h[j] * Wo[j * 17 + a];
  Out[idx] = acc;
}

// ---------------------------------------------------------------------------
extern "C" void kernel_launch(void* const* d_in, const int* in_sizes, int n_in,
                              void* d_out, int out_size, void* d_ws, size_t ws_size,
                              hipStream_t stream) {
  const float* x      = (const float*)d_in[0];
  const float* w_in   = (const float*)d_in[1];
  const float* b_in   = (const float*)d_in[2];
  const float* in_w   = (const float*)d_in[3];
  const float* in_b   = (const float*)d_in[4];
  const float* conv_w = (const float*)d_in[5];
  const float* conv_b = (const float*)d_in[6];
  const float* xp_w   = (const float*)d_in[7];
  const float* dt_w   = (const float*)d_in[8];
  const float* dt_b   = (const float*)d_in[9];
  const float* A_log  = (const float*)d_in[10];
  const float* D_par  = (const float*)d_in[11];
  const float* y1_w   = (const float*)d_in[12];
  const float* y1_b   = (const float*)d_in[13];
  const float* out_w  = (const float*)d_in[14];
  const float* out_b  = (const float*)d_in[15];
  const float* ln_s   = (const float*)d_in[16];
  const float* ln_b   = (const float*)d_in[17];
  const float* w_out  = (const float*)d_in[18];
  const float* b_out  = (const float*)d_in[19];
  float* out = (float*)d_out;

  // ---- workspace carve-up (~105 MB, buffers reused across stages) ----
  char* ws = (char*)d_ws;
  size_t off = 0;
  auto allocf = [&](size_t n) { float* p = (float*)(ws + off);
                                off += ((n * 4 + 255) / 256) * 256; return p; };
  auto allocb = [&](size_t n) { bf16_t* p = (bf16_t*)(ws + off);
                                off += ((n * 2 + 255) / 256) * 256; return p; };
  const int T = T_TOKENS;
  float* h     = allocf((size_t)T * DM);    // layer io
  float* xn    = allocf((size_t)T * DM);    // rmsnorm out, later 'o'
  float* xr    = allocf((size_t)T * 1024);  // [xp | res]
  float* u     = allocf((size_t)T * DI);    // conv+silu out, later 'z'
  float* proj  = allocf((size_t)T * 48);
  float* delta = allocf((size_t)T * DI);    // later y1 out
  float* yg    = allocf((size_t)T * DI);    // gated scan out
  bf16_t* w_inT  = allocb(256 * 128);
  bf16_t* in_wT  = allocb((size_t)4 * 1024 * 256);
  bf16_t* xp_wT  = allocb((size_t)4 * 48 * 512);
  bf16_t* y1_wT  = allocb((size_t)4 * 512 * 512);
  bf16_t* out_wT = allocb((size_t)4 * 256 * 512);
  (void)allocb(64 * 512);                   // pad: N-edge over-read stays in ws

  // ---- pack all weights to bf16 transposed (N x K) ----
  auto pack = [&](const float* W, bf16_t* Wt, int K, int N) {
    int n = K * N;
    packT_kernel<<<(n + 255) / 256, 256, 0, stream>>>(W, Wt, K, N);
  };
  pack(w_in, w_inT, 128, 256);
  for (int l = 0; l < 4; ++l) {
    pack(in_w  + (size_t)l * 256 * 1024, in_wT  + (size_t)l * 1024 * 256, 256, 1024);
    pack(xp_w  + (size_t)l * 512 * 48,   xp_wT  + (size_t)l * 48 * 512,   512, 48);
    pack(y1_w  + (size_t)l * 512 * 512,  y1_wT  + (size_t)l * 512 * 512,  512, 512);
    pack(out_w + (size_t)l * 512 * 256,  out_wT + (size_t)l * 256 * 512,  512, 256);
  }

  auto gemm = [&](const float* A, int lda, const bf16_t* Wt, const float* bias,
                  float* C, int ldc, int M, int N, int K) {
    dim3 grid((M + BM - 1) / BM, (N + BN - 1) / BN);
    gemm_bf16_wmma<<<grid, 128, 0, stream>>>(A, lda, Wt, bias, C, ldc, M, N, K);
  };

  // ---- input projection: h = x @ w_in + b_in ----
  gemm(x, 128, w_inT, b_in, h, DM, T, DM, 128);

  for (int l = 0; l < 4; ++l) {
    // xn = rmsnorm(h)
    rmsnorm_kernel<DM><<<T, 256, 0, stream>>>(h, DM, nullptr, 0, xn, DM);
    // xr = xn @ in_w + in_b   (1024 = [xp | res])
    gemm(xn, DM, in_wT + (size_t)l * 1024 * 256, in_b + l * 1024, xr, 1024, T, 1024, DM);
    // u = silu(depthwise_conv(xp) + cb)
    conv_silu_kernel<<<(T * DI + 255) / 256, 256, 0, stream>>>(
        xr, 1024, conv_w + (size_t)l * 4 * DI, conv_b + l * DI, u);
    // proj = xp @ xp_w   (N = 48)
    gemm(xr, 1024, xp_wT + (size_t)l * 48 * 512, nullptr, proj, 48, T, 48, DI);
    // delta = softplus(proj[:, :16] @ dt_w + dt_b)
    dtproj_kernel<<<(T * DI + 255) / 256, 256, 0, stream>>>(
        proj, dt_w + (size_t)l * DTR * DI, dt_b + l * DI, delta);
    // yg = selective_scan(u, delta, B, C) * silu(res)
    scan_kernel<<<128, 256, 0, stream>>>(
        u, delta, proj, A_log + (size_t)l * DI * NST, D_par + l * DI,
        xr + DI, 1024, yg);
    // y1 = yg @ y1_w + y1_b   (into delta buffer)
    gemm(yg, DI, y1_wT + (size_t)l * 512 * 512, y1_b + l * DI, delta, DI, T, DI, DI);
    // z = rmsnorm(xp + y1)    (into u buffer)
    rmsnorm_kernel<DI><<<T, 256, 0, stream>>>(delta, DI, xr, 1024, u, DI);
    // o = z @ out_w + out_b   (into xn buffer)
    gemm(u, DI, out_wT + (size_t)l * 256 * 512, out_b + l * DM, xn, DM, T, DM, DI);
    // h = layernorm(softmax(o))
    smax_ln_kernel<<<T, 256, 0, stream>>>(xn, ln_s + l * DM, ln_b + l * DM, h);
  }

  // ---- logits = h @ w_out + b_out ----
  head_kernel<<<(T * 17 + 255) / 256, 256, 0, stream>>>(h, w_out, b_out, out);
}
